// GCN_4587025072673
// MI455X (gfx1250) — matrix operand order repaired and checked by reading, not compiled, not run on hardware
//
#include <hip/hip_runtime.h>

typedef _Float16 half_t;
typedef _Float16 v16h __attribute__((ext_vector_type(16)));
typedef _Float16 v8h  __attribute__((ext_vector_type(8)));
typedef float    v8f  __attribute__((ext_vector_type(8)));
typedef float    v4f  __attribute__((ext_vector_type(4)));

#define GB 8
#define GN 2048
#define GD 128
#define GF 32
#define GL 3

#define BLOCK_M 64            // rows per workgroup (4 waves x 16 rows)
#define THREADS 128           // 4 wave32s
#define KB 32                 // K per WMMA step (16x16x32 f16)

// ---------------- f32 -> f16 bulk convert (8 elems/thread, b128 loads) ----------------
__global__ __launch_bounds__(256) void cvt8_kernel(const float* __restrict__ src,
                                                   half_t* __restrict__ dst, int n8) {
  int i = blockIdx.x * blockDim.x + threadIdx.x;
  if (i < n8) {
    v4f a = ((const v4f*)src)[2 * i];
    v4f b = ((const v4f*)src)[2 * i + 1];
    union { v8h v; half_t e[8]; } o;
#pragma unroll
    for (int j = 0; j < 4; ++j) { o.e[j] = (half_t)a[j]; o.e[4 + j] = (half_t)b[j]; }
    ((v8h*)dst)[i] = o.v;
  }
}

// ---------------- f32 MxN row-major -> f16 NxM (transpose + convert) ----------------
__global__ __launch_bounds__(256) void cvtT_kernel(const float* __restrict__ src,
                                                   half_t* __restrict__ dst, int M, int N) {
  int i = blockIdx.x * blockDim.x + threadIdx.x;
  if (i < M * N) {
    int m = i / N, n = i - m * N;
    dst[(size_t)n * M + m] = (half_t)src[i];
  }
}

// ---------------- fused WMMA GEMM: C = act(A @ B + bias) [* mask] ----------------
// A : MxK row-major, lda == K (f32 converted in-register, or f16).
// BT: B stored column-major [NOUT][K] f16 -> every B-fragment is two 16B global
//     runs; all tile strides are compile-time IOFFSET immediates (K is a template arg).
// Grid: x = M blocks, y = batch, z = column block (BN columns per workgroup, NOUT total).
// OUT_MODE: 0 = f16 row-major [M][NOUT]; 1 = f16 transposed per-batch [NOUT][GN];
//           2 = f32 row-major [M][NOUT].
template <int BN, int NOUT, int K, bool A_IS_F32, bool RELU, int OUT_MODE, bool MASK>
__global__ __launch_bounds__(THREADS) void wmma_gemm_kernel(
    const void* __restrict__ Aptr, const half_t* __restrict__ BT,
    const float* __restrict__ bias, const float* __restrict__ maskp,
    void* __restrict__ Cptr,
    unsigned long long aBatch, unsigned long long bBatch, unsigned long long cBatch) {
  constexpr int NT = BN / 16;
  static_assert(K % (2 * KB) == 0, "K must be a multiple of 64");

  const int tid  = threadIdx.x;
  const int wave = tid >> 5;
  const int lane = tid & 31;
  const int hi   = lane >> 4;          // K-half select (WMMA operand layout)
  const int mrow = lane & 15;
  const long long row = (long long)blockIdx.x * BLOCK_M + wave * 16 + mrow;

  const float*  Af = (const float*)Aptr  + (A_IS_F32 ? blockIdx.y * aBatch : 0ull);
  const half_t* Ah = (const half_t*)Aptr + (A_IS_F32 ? 0ull : blockIdx.y * aBatch);
  const half_t* Bb = BT + blockIdx.y * bBatch + (unsigned long long)blockIdx.z * BN * K;

  // per-lane base pointers; advanced by 2*KB elements per loop iteration
  const float*  aptr_f = Af + row * (long long)K + hi * 8;
  const half_t* aptr_h = Ah + row * (long long)K + hi * 8;
  const half_t* bptr   = Bb + (long long)mrow * K + hi * 8;

  v8f acc[NT] = {};

  struct Frag { v16h a; v16h b[NT]; };

  // koff is a compile-time constant at every call site -> immediate offsets only
  auto loadFrag = [&](int koff) {
    Frag f;
    if constexpr (A_IS_F32) {
      v4f f0 = *(const v4f*)(aptr_f + koff);
      v4f f1 = *(const v4f*)(aptr_f + koff + 4);
      v4f f2 = *(const v4f*)(aptr_f + koff + 16);
      v4f f3 = *(const v4f*)(aptr_f + koff + 20);
      union { v16h v; half_t e[16]; } u;
#pragma unroll
      for (int j = 0; j < 4; ++j) {
        u.e[j]      = (half_t)f0[j];
        u.e[4 + j]  = (half_t)f1[j];
        u.e[8 + j]  = (half_t)f2[j];
        u.e[12 + j] = (half_t)f3[j];
      }
      f.a = u.v;
    } else {
      union { v16h v; v8h h2[2]; } u;
      u.h2[0] = *(const v8h*)(aptr_h + koff);
      u.h2[1] = *(const v8h*)(aptr_h + koff + 16);
      f.a = u.v;
    }
#pragma unroll
    for (int t = 0; t < NT; ++t) {
      union { v16h v; v8h h2[2]; } u;
      u.h2[0] = *(const v8h*)(bptr + t * 16 * K + koff);
      u.h2[1] = *(const v8h*)(bptr + t * 16 * K + koff + 16);
      f.b[t] = u.v;
    }
    return f;
  };

  auto mma = [&](const Frag& f) {
#pragma unroll
    for (int t = 0; t < NT; ++t)
      acc[t] = __builtin_amdgcn_wmma_f32_16x16x32_f16(
          false, f.a, false, f.b[t], (short)0, acc[t], false, false);
  };

  // software pipeline: prologue loads, branch-free body, peeled tail
  Frag fa = loadFrag(0);
  Frag fb = loadFrag(KB);
#pragma unroll 1
  for (int it = 0; it < K / (2 * KB) - 1; ++it) {
    if constexpr (A_IS_F32) __builtin_prefetch((const void*)(aptr_f + 8 * KB), 0, 0);
    mma(fa);
    fa = loadFrag(2 * KB);
    mma(fb);
    fb = loadFrag(3 * KB);
    aptr_f += 2 * KB;
    aptr_h += 2 * KB;
    bptr   += 2 * KB;
  }
  mma(fa);
  mma(fb);

  // ---- epilogue: bias, ReLU, mask, store ----
#pragma unroll
  for (int t = 0; t < NT; ++t) {
    int colg = blockIdx.z * BN + t * 16 + mrow;    // global output column
    float bv = bias ? bias[colg] : 0.0f;
    if constexpr (OUT_MODE == 1) {
      // transposed f16 output: C[b][colg][n..n+7] — one b128 store per tile per lane
      long long rbase = (long long)blockIdx.x * BLOCK_M + wave * 16 + hi * 8;
      long long bidx  = rbase / GN;           // batch (M is batch-folded, BLOCK_M | GN)
      long long n     = rbase - bidx * GN;
      half_t* Ct = (half_t*)Cptr + bidx * cBatch + (long long)colg * GN + n;
      union { v8h v; half_t e[8]; } o;
#pragma unroll
      for (int a = 0; a < 8; ++a) {
        float v = acc[t][a] + bv;
        if constexpr (RELU) v = v > 0.0f ? v : 0.0f;
        o.e[a] = (half_t)v;
      }
      *(v8h*)Ct = o.v;
    } else {
#pragma unroll
      for (int a = 0; a < 8; ++a) {
        long long r = (long long)blockIdx.x * BLOCK_M + wave * 16 + (hi ? 8 + a : a);
        float v = acc[t][a] + bv;
        if constexpr (RELU) v = v > 0.0f ? v : 0.0f;
        if constexpr (MASK) v *= maskp[r];
        if constexpr (OUT_MODE == 0)
          ((half_t*)Cptr + blockIdx.y * cBatch)[r * (long long)NOUT + colg] = (half_t)v;
        else
          ((float*)Cptr)[r * (long long)NOUT + colg] = v;
      }
    }
  }
}

extern "C" void kernel_launch(void* const* d_in, const int* in_sizes, int n_in,
                              void* d_out, int out_size, void* d_ws, size_t ws_size,
                              hipStream_t stream) {
  (void)in_sizes; (void)n_in; (void)out_size; (void)ws_size;
  const float* x    = (const float*)d_in[0];   // [8,2048,128]
  const float* adj  = (const float*)d_in[1];   // [8,2048,2048]
  const float* mask = (const float*)d_in[2];   // [8,2048,1]
  const float* W    = (const float*)d_in[3];   // [3,128,128]
  const float* bvec = (const float*)d_in[4];   // [3,128]
  const float* Wout = (const float*)d_in[5];   // [128,32]
  const float* bout = (const float*)d_in[6];   // [32]

  half_t* h    = (half_t*)d_ws;                      // [8*2048,128] row-major f16
  half_t* msgT = h    + (size_t)GB * GN * GD;        // [8][128][2048] transposed f16
  half_t* Wt   = msgT + (size_t)GB * GN * GD;        // [3][128out][128in] transposed f16
  half_t* WoT  = Wt   + (size_t)GL * GD * GD;        // [32][128] transposed f16

  // one-time conversions
  {
    int n8 = GB * GN * GD / 8;
    cvt8_kernel<<<(n8 + 255) / 256, 256, 0, stream>>>(x, h, n8);
    for (int i = 0; i < GL; ++i)
      cvtT_kernel<<<(GD * GD + 255) / 256, 256, 0, stream>>>(
          W + (size_t)i * GD * GD, Wt + (size_t)i * GD * GD, GD, GD);
    cvtT_kernel<<<(GD * GF + 255) / 256, 256, 0, stream>>>(Wout, WoT, GD, GF);
  }

  for (int i = 0; i < GL; ++i) {
    // msgT = (h @ W[i] + b[i])^T  (M = 16384, K = 128; BN=32 x 4 col-blocks: low
    // register pressure -> high occupancy for this latency-bound small GEMM)
    wmma_gemm_kernel<32, GD, GD, false, false, 1, false>
        <<<dim3(GB * GN / BLOCK_M, 1, GD / 32), THREADS, 0, stream>>>(
            h, Wt + (size_t)i * GD * GD, bvec + i * GD, nullptr, msgT,
            0ull, 0ull, (unsigned long long)GD * GN);
    // h = relu(adj @ msg)  (per batch: M = 2048, K = 2048; BN=128 so each streamed
    // adj element feeds all 128 output columns -> adj read exactly once per layer)
    wmma_gemm_kernel<GD, GD, GN, true, true, 0, false>
        <<<dim3(GN / BLOCK_M, GB, 1), THREADS, 0, stream>>>(
            adj, msgT, nullptr, nullptr, h,
            (unsigned long long)GN * GN, (unsigned long long)GD * GN,
            (unsigned long long)GN * GD);
  }

  // type_pred = (h @ W_out + b_out) * node_mask   (f32 out, NOUT = 32)
  wmma_gemm_kernel<GF, GF, GD, false, false, 2, true>
      <<<dim3(GB * GN / BLOCK_M, 1, 1), THREADS, 0, stream>>>(
          h, WoT, bout, mask, d_out,
          0ull, 0ull, 0ull);
}
